// PointerGenerator_29463475651505
// MI455X (gfx1250) — matrix operand order repaired and checked by reading, not compiled, not run on hardware
//
#include <hip/hip_runtime.h>
#include <hip/hip_bf16.h>

typedef __attribute__((ext_vector_type(16))) __bf16 v16bf;
typedef __attribute__((ext_vector_type(8)))  float  v8f;

#define BSZ   32
#define SRCL  400
#define NVOC  50257
#define DEMB  512
#define D3    1536

#define WMMA_BF16(A,B,C) __builtin_amdgcn_wmma_f32_16x16x32_bf16(false,(A),false,(B),(short)0,(C),false,false)

static __device__ __forceinline__ unsigned ordf(float f) {
  unsigned u = __float_as_uint(f);
  return (u & 0x80000000u) ? ~u : (u | 0x80000000u);
}
static __device__ __forceinline__ float unordf(unsigned u) {
  return __uint_as_float((u & 0x80000000u) ? (u ^ 0x80000000u) : ~u);
}

// Build a bf16 A-fragment (16x32, per-lane row = lane&15, one 16-elem K run)
// directly from an fp32 row in global memory. kb already includes the
// lane-half (+16) offset, so the run is [kb..kb+15].
static __device__ __forceinline__ v16bf load_a_from_f32(const float* __restrict__ row, int kb) {
  const float4 x0 = *(const float4*)(row + kb);
  const float4 x1 = *(const float4*)(row + kb + 4);
  const float4 y0 = *(const float4*)(row + kb + 8);
  const float4 y1 = *(const float4*)(row + kb + 12);
  v16bf r;
  r[0]=(__bf16)x0.x; r[1]=(__bf16)x0.y; r[2]=(__bf16)x0.z; r[3]=(__bf16)x0.w;
  r[4]=(__bf16)x1.x; r[5]=(__bf16)x1.y; r[6]=(__bf16)x1.z; r[7]=(__bf16)x1.w;
  r[8]=(__bf16)y0.x; r[9]=(__bf16)y0.y; r[10]=(__bf16)y0.z; r[11]=(__bf16)y0.w;
  r[12]=(__bf16)y1.x; r[13]=(__bf16)y1.y; r[14]=(__bf16)y1.z; r[15]=(__bf16)y1.w;
  return r;
}

// ---------------- prep kernels ----------------

__global__ void k_init(float* __restrict__ P1, unsigned* __restrict__ gmax, float* __restrict__ gsum) {
  size_t n = (size_t)NVOC * 32;
  for (size_t i = (size_t)blockIdx.x * blockDim.x + threadIdx.x; i < n;
       i += (size_t)gridDim.x * blockDim.x)
    P1[i] = 0.f;
  if (blockIdx.x == 0 && threadIdx.x < 32) {
    gmax[threadIdx.x] = 0x007FFFFFu;   // ordf(-inf)
    gsum[threadIdx.x] = 0.f;
  }
}

__global__ void k_ccat(const float* __restrict__ hd, const float* __restrict__ ce,
                       const float* __restrict__ cd, __bf16* __restrict__ ccbf) {
  int i = blockIdx.x * blockDim.x + threadIdx.x;
  if (i >= BSZ * D3) return;
  int b = i / D3, j = i - b * D3;
  float v = (j < 512) ? hd[b * 512 + j]
          : (j < 1024) ? ce[b * 512 + (j - 512)]
                       : cd[b * 512 + (j - 1024)];
  ccbf[i] = (__bf16)v;
}

__global__ void k_wpt(const float* __restrict__ Wp, __bf16* __restrict__ WpT) {
  int i = blockIdx.x * blockDim.x + threadIdx.x;
  if (i >= DEMB * D3) return;
  int k = i / D3, n = i - k * D3;
  WpT[(size_t)n * DEMB + k] = (__bf16)Wp[i];
}

__global__ void k_scatter(const int* __restrict__ x, const float* __restrict__ ae,
                          float* __restrict__ P1) {
  int i = blockIdx.x * blockDim.x + threadIdx.x;
  if (i >= BSZ * SRCL) return;
  int b = i / SRCL, s = i - b * SRCL;
  int id = x[(size_t)b * SRCL + s];
  if (id >= 0 && id < NVOC) P1[(size_t)id * 32 + b] = ae[(size_t)b * SRCL + s];
}

// ---------------- main fused WMMA kernel ----------------
// 2 waves / block, 16 vocab rows per wave. Per wave:
//   T = tanh(W_emb[16 x 512] @ W_proj[:, j:j+16])     (bf16 WMMA, K=512)
//   logits0 += T @ c_cat[:, j:j+32]^T                 (bf16 WMMA via LDS transpose)
//   logits_u += W_u[16, j:j+32] @ c_cat^T             (bf16 WMMA, A from global)
__global__ __launch_bounds__(64) void k_main(
    const float* __restrict__ Wemb, const float* __restrict__ Wu,
    const __bf16* __restrict__ WpT, const __bf16* __restrict__ ccbf,
    const float* __restrict__ bout, const float* __restrict__ bu,
    float* __restrict__ L0, float* __restrict__ PU) {
  __shared__ __bf16 sA[2][16][DEMB];  // 32 KB: staged bf16 W_emb tiles
  __shared__ __bf16 sT[2][16][32];    //  2 KB: tanh(T) staging for GEMM-2 A frags

  const int tid  = threadIdx.x;
  const int w    = tid >> 5;
  const int lane = tid & 31;
  const int half = lane >> 4;
  const int l15  = lane & 15;
  const int vbase = blockIdx.x * 32 + w * 16;

  // Stage W_emb rows into LDS as bf16 (each wave fills its own region).
  for (int idx = lane; idx < 16 * (DEMB / 4); idx += 32) {
    int r = idx >> 7;              // DEMB/4 == 128 float4 per row
    int c = (idx & 127) << 2;
    int vr = vbase + r; if (vr >= NVOC) vr = NVOC - 1;
    float4 f = *(const float4*)(Wemb + (size_t)vr * DEMB + c);
    sA[w][r][c + 0] = (__bf16)f.x; sA[w][r][c + 1] = (__bf16)f.y;
    sA[w][r][c + 2] = (__bf16)f.z; sA[w][r][c + 3] = (__bf16)f.w;
  }
  asm volatile("s_wait_dscnt 0" ::: "memory");  // wave-local LDS RAW

  int vrow = vbase + l15; if (vrow >= NVOC) vrow = NVOC - 1;
  const float* wuRow = Wu + (size_t)vrow * D3;

  v8f accL = {}, accH = {}, uaL = {}, uaH = {};

  for (int jc = 0; jc < D3 / 16; ++jc) {           // 96 j-chunks
    const int j0 = jc * 16;
    v8f t = {};
#pragma unroll
    for (int kc = 0; kc < DEMB / 32; ++kc) {       // 16 WMMAs over K=512
      v16bf a;
      const int kb = kc * 32 + half * 16;
#pragma unroll
      for (int e = 0; e < 8; ++e) {
        a[e]     = sA[w][l15][kb + e];
        a[8 + e] = sA[w][l15][kb + 8 + e];
      }
      v16bf b = *(const v16bf*)(WpT + (size_t)(j0 + l15) * DEMB + kc * 32 + half * 16);
      t = WMMA_BF16(a, b, t);
    }
    // tanh epilogue of GEMM-1, stash as bf16 in A-fragment-friendly LDS tile
    {
      const int mrow = half * 8;
      const int jcol = (jc & 1) * 16 + l15;
#pragma unroll
      for (int r = 0; r < 8; ++r)
        sT[w][mrow + r][jcol] = (__bf16)tanhf(t[r]);
    }
    if (jc & 1) {
      asm volatile("s_wait_dscnt 0" ::: "memory");
      const int k0 = (jc >> 1) * 32;
      // A2: transposed tanh(T) block from LDS
      v16bf a2;
      const int kb2 = half * 16;
#pragma unroll
      for (int e = 0; e < 8; ++e) {
        a2[e]     = sT[w][l15][kb2 + e];
        a2[8 + e] = sT[w][l15][kb2 + 8 + e];
      }
      // B: c_cat columns, batches 0-15 and 16-31
      v16bf bL = *(const v16bf*)(ccbf + (size_t)l15 * D3 + k0 + half * 16);
      v16bf bH = *(const v16bf*)(ccbf + (size_t)(16 + l15) * D3 + k0 + half * 16);
      accL = WMMA_BF16(a2, bL, accL);
      accH = WMMA_BF16(a2, bH, accH);
      // copy-gate path: A straight from W_u (fp32 -> bf16 in flight)
      v16bf au = load_a_from_f32(wuRow, k0 + half * 16);
      uaL = WMMA_BF16(au, bL, uaL);
      uaH = WMMA_BF16(au, bH, uaH);
      if (k0 + 160 < D3) __builtin_prefetch(wuRow + k0 + 128, 0, 1);
    }
  }

  // Epilogue: C/D layout is N = lane&15, M = r + 8*(lane>=16)
  const int mrow = half * 8;
#pragma unroll
  for (int r = 0; r < 8; ++r) {
    int v = vbase + mrow + r;
    if (v < NVOC) {
      float bo = bout[v], buv = bu[v];
      L0[(size_t)v * 32 + l15]      = accL[r] + bo;
      L0[(size_t)v * 32 + 16 + l15] = accH[r] + bo;
      PU[(size_t)v * 32 + l15]      = 1.f / (1.f + __expf(-(uaL[r] + buv)));
      PU[(size_t)v * 32 + 16 + l15] = 1.f / (1.f + __expf(-(uaH[r] + buv)));
    }
  }
}

// ---------------- softmax reductions ----------------

__global__ void k_max(const float* __restrict__ L0, unsigned* __restrict__ gmax) {
  __shared__ float sm[8][32];
  int b = threadIdx.x & 31, slot = threadIdx.x >> 5;
  int vend = (blockIdx.x + 1) * 512; if (vend > NVOC) vend = NVOC;
  float m = -__builtin_huge_valf();
  for (int v = blockIdx.x * 512 + slot; v < vend; v += 8)
    m = fmaxf(m, L0[(size_t)v * 32 + b]);
  sm[slot][b] = m;
  __syncthreads();
  if (slot == 0) {
#pragma unroll
    for (int i = 1; i < 8; ++i) m = fmaxf(m, sm[i][b]);
    atomicMax(&gmax[b], ordf(m));
  }
}

__global__ void k_sum(const float* __restrict__ L0, const unsigned* __restrict__ gmax,
                      float* __restrict__ gsum) {
  __shared__ float sm[8][32];
  int b = threadIdx.x & 31, slot = threadIdx.x >> 5;
  float mx = unordf(gmax[b]);
  int vend = (blockIdx.x + 1) * 512; if (vend > NVOC) vend = NVOC;
  float s = 0.f;
  for (int v = blockIdx.x * 512 + slot; v < vend; v += 8)
    s += __expf(L0[(size_t)v * 32 + b] - mx);
  sm[slot][b] = s;
  __syncthreads();
  if (slot == 0) {
#pragma unroll
    for (int i = 1; i < 8; ++i) s += sm[i][b];
    atomicAdd(&gsum[b], s);
  }
}

// ---------------- final blend (LDS transpose [v][b] -> [b][v]) ----------------

__global__ __launch_bounds__(1024) void k_final(
    const float* __restrict__ L0, const float* __restrict__ PU,
    const float* __restrict__ P1, const unsigned* __restrict__ gmax,
    const float* __restrict__ gsum, float* __restrict__ out) {
  __shared__ float s[32][33];
  int x = threadIdx.x, y = threadIdx.y;
  int v0 = blockIdx.x * 32;
  int v = v0 + y;
  float val = 0.f;
  if (v < NVOC) {
    float mx = unordf(gmax[x]);
    float p0 = __expf(L0[(size_t)v * 32 + x] - mx) / gsum[x];
    float pu = PU[(size_t)v * 32 + x];
    float p1 = P1[(size_t)v * 32 + x];
    val = pu * p1 + (1.f - pu) * p0;
  }
  s[y][x] = val;
  __syncthreads();
  int vv = v0 + x;
  if (vv < NVOC) out[(size_t)y * NVOC + vv] = s[x][y];
}

// ---------------- launcher ----------------

extern "C" void kernel_launch(void* const* d_in, const int* in_sizes, int n_in,
                              void* d_out, int out_size, void* d_ws, size_t ws_size,
                              hipStream_t stream) {
  const int*   x    = (const int*)d_in[0];
  const float* ae   = (const float*)d_in[1];
  const float* ce   = (const float*)d_in[2];
  const float* hd   = (const float*)d_in[3];
  const float* cd   = (const float*)d_in[4];
  const float* Wemb = (const float*)d_in[5];
  const float* Wp   = (const float*)d_in[6];
  const float* bout = (const float*)d_in[7];
  const float* Wu   = (const float*)d_in[8];
  const float* bu   = (const float*)d_in[9];
  float* out = (float*)d_out;

  uintptr_t base = (uintptr_t)d_ws;
  size_t off = 0;
  auto take = [&](size_t bytes) -> void* {
    void* p = (void*)(base + off);
    off = (off + bytes + 255) & ~(size_t)255;
    return p;
  };
  __bf16*   ccbf = (__bf16*)take((size_t)BSZ * D3 * 2);
  __bf16*   WpT  = (__bf16*)take((size_t)DEMB * D3 * 2);
  float*    L0   = (float*)take((size_t)NVOC * 32 * 4);
  float*    PU   = (float*)take((size_t)NVOC * 32 * 4);
  float*    P1   = (float*)take((size_t)NVOC * 32 * 4);
  unsigned* gmax = (unsigned*)take(32 * 4);
  float*    gsum = (float*)take(32 * 4);

  k_init<<<512, 256, 0, stream>>>(P1, gmax, gsum);
  k_ccat<<<(BSZ * D3 + 255) / 256, 256, 0, stream>>>(hd, ce, cd, ccbf);
  k_wpt<<<(DEMB * D3 + 255) / 256, 256, 0, stream>>>(Wp, WpT);
  k_scatter<<<(BSZ * SRCL + 255) / 256, 256, 0, stream>>>(x, ae, P1);
  k_main<<<(NVOC + 31) / 32, 64, 0, stream>>>(Wemb, Wu, WpT, ccbf, bout, bu, L0, PU);
  k_max<<<(NVOC + 511) / 512, 256, 0, stream>>>(L0, gmax);
  k_sum<<<(NVOC + 511) / 512, 256, 0, stream>>>(L0, gmax, gsum);
  k_final<<<(NVOC + 31) / 32, dim3(32, 32), 0, stream>>>(L0, PU, P1, gmax, gsum, out);
}
